// MoELayer_27152783245926
// MI455X (gfx1250) — matrix operand order repaired and checked by reading, not compiled, not run on hardware
//
#include <hip/hip_runtime.h>

// ---------------- model dims ----------------
constexpr int kD   = 768;
constexpr int kH   = 12;
constexpr int kDH  = 64;
constexpr int kDFF = 3072;
constexpr int kE   = 8;
constexpr int kS   = 2048;

// ---------------- WMMA types ----------------
typedef __attribute__((ext_vector_type(16))) __bf16 v16bf;
typedef __attribute__((ext_vector_type(8)))  __bf16 v8bf;
typedef __attribute__((ext_vector_type(8)))  float  v8f;

__device__ __forceinline__ unsigned short f2bf(float x) {
    union { float f; unsigned u; } c; c.f = x;
    unsigned u = c.u;
    unsigned r = (u + 0x7fffu + ((u >> 16) & 1u)) >> 16;   // RNE
    return (unsigned short)r;
}
__device__ __forceinline__ float bf2f(unsigned short u) {
    union { unsigned u; float f; } c; c.u = ((unsigned)u) << 16; return c.f;
}

__device__ __forceinline__ v16bf make_frag(v8bf lo, v8bf hi) {
    v16bf a;
#pragma unroll
    for (int i = 0; i < 8; ++i) { a[i] = lo[i]; a[i + 8] = hi[i]; }
    return a;
}

// A fragment (16x32 bf16, M x K). rowptr = &A[row_of_this_lane][ktile]
// lane<16: K 0..7 and 16..23 ; lane>=16: K 8..15 and 24..31
__device__ __forceinline__ v16bf load_a(const unsigned short* rowptr) {
    const int lane = threadIdx.x & 31;
    const int k0 = (lane < 16) ? 0 : 8;
    v8bf lo = *(const v8bf*)(rowptr + k0);
    v8bf hi = *(const v8bf*)(rowptr + k0 + 16);
    return make_frag(lo, hi);
}

// B fragment (32x16, K x N) with B[k][n] = W[n][k]. colptr = &W[col_of_this_lane][ktile]
// lane<16: K 0..7 and 16..23 ; lane>=16: K 8..15 and 24..31
__device__ __forceinline__ v16bf load_b(const unsigned short* colptr) {
    const int lane = threadIdx.x & 31;
    const int k0 = (lane < 16) ? 0 : 8;
    v8bf lo = *(const v8bf*)(colptr + k0);
    v8bf hi = *(const v8bf*)(colptr + k0 + 16);
    return make_frag(lo, hi);
}

__device__ __forceinline__ v8f wmma_bf16(v16bf a, v16bf b, v8f c) {
    return __builtin_amdgcn_wmma_f32_16x16x32_bf16(false, a, false, b, (short)0, c, false, false);
}

// ---------------- CDNA5 async global->LDS staging ----------------
__device__ __forceinline__ void async_ld16(unsigned lds_off, const unsigned short* g) {
    asm volatile("global_load_async_to_lds_b128 %0, %1, off"
                 :: "v"(lds_off), "v"(g) : "memory");
}
__device__ __forceinline__ void wait_async0() {
    asm volatile("s_wait_asynccnt 0" ::: "memory");
}
__device__ __forceinline__ void wait_async2() {
    asm volatile("s_wait_asynccnt 2" ::: "memory");
}

// Stage a 64(N) x 32(K) bf16 tile of W (row-major [N,K]) into LDS.
// 128 threads, 2 x 16B chunks each -> 2 async instructions per wave.
__device__ __forceinline__ void stage_B_async(const unsigned short* Wtile, // W + n0*K + kt
                                              int K, unsigned lbase, int tid) {
#pragma unroll
    for (int i = 0; i < 2; ++i) {
        const int c = tid + i * 128;               // chunk 0..255
        const int col = c >> 2, part = (c & 3) * 8;
        async_ld16(lbase + (unsigned)(col * 32 + part) * 2,
                   Wtile + (size_t)col * K + part);
    }
}

// ---------------- elementwise / setup kernels ----------------
__global__ void cvt_bf16_kernel(const float* __restrict__ in,
                                unsigned short* __restrict__ out, int n) {
    int i = blockIdx.x * blockDim.x + threadIdx.x;
    int stride = gridDim.x * blockDim.x;
    for (; i < n; i += stride) out[i] = f2bf(in[i]);
}

__global__ void copy_f32_kernel(const float* __restrict__ in,
                                float* __restrict__ out, int n) {
    int i = blockIdx.x * blockDim.x + threadIdx.x;
    int stride = gridDim.x * blockDim.x;
    for (; i < n; i += stride) out[i] = in[i];
}

// LayerNorm over D=768, one block (256 threads) per token, bf16 output.
__global__ void __launch_bounds__(256) ln_bf16_kernel(const float* __restrict__ x,
                                                      const float* __restrict__ g,
                                                      const float* __restrict__ b,
                                                      unsigned short* __restrict__ out) {
    __shared__ float red[256];
    const int t = blockIdx.x;
    const float* xr = x + (size_t)t * kD;
    float s = 0.f;
    for (int i = threadIdx.x; i < kD; i += 256) s += xr[i];
    red[threadIdx.x] = s; __syncthreads();
    for (int o = 128; o > 0; o >>= 1) {
        if (threadIdx.x < o) red[threadIdx.x] += red[threadIdx.x + o];
        __syncthreads();
    }
    const float mu = red[0] / kD;
    __syncthreads();
    float v = 0.f;
    for (int i = threadIdx.x; i < kD; i += 256) { float d = xr[i] - mu; v += d * d; }
    red[threadIdx.x] = v; __syncthreads();
    for (int o = 128; o > 0; o >>= 1) {
        if (threadIdx.x < o) red[threadIdx.x] += red[threadIdx.x + o];
        __syncthreads();
    }
    const float rstd = rsqrtf(red[0] / kD + 1e-5f);
    for (int i = threadIdx.x; i < kD; i += 256)
        out[(size_t)t * kD + i] = f2bf((xr[i] - mu) * rstd * g[i] + b[i]);
}

// v [S, H*DH] bf16 -> vT [H*DH, S] bf16
__global__ void transpose_v_kernel(const unsigned short* __restrict__ v,
                                   unsigned short* __restrict__ vt) {
    int idx = blockIdx.x * blockDim.x + threadIdx.x;
    if (idx >= kS * kD) return;
    int s = idx / kD, c = idx % kD;
    vt[(size_t)c * kS + s] = v[idx];
}

// ---------------- tiled GEMM: 128 threads = 4 waves, 64x64 block tile ----------------
// B (weights) double-buffer staged through LDS via async loads; A read direct.
// C[M,N](bf16) = A[M,K] @ W[N,K]^T + bias
__global__ void __launch_bounds__(128) gemm_bf16_kernel(const unsigned short* __restrict__ A,
                                                        const unsigned short* __restrict__ W,
                                                        const float* __restrict__ bias,
                                                        unsigned short* __restrict__ C,
                                                        int N, int K) {
    __shared__ __align__(16) unsigned short Bs[2][64 * 32];
    const int tid = threadIdx.x;
    const int wave = tid >> 5, lane = tid & 31;
    const int nl = lane & 15;
    const int m0 = blockIdx.y * 64 + wave * 16;
    const int n0 = blockIdx.x * 64;
    const unsigned short* ap = A + (size_t)(m0 + nl) * K;
    const unsigned short* wt = W + (size_t)n0 * K;
    v8f acc[4] = {};
    stage_B_async(wt, K, (unsigned)(size_t)&Bs[0][0], tid);
    int buf = 0;
    for (int kt = 0; kt < K; kt += 32, buf ^= 1) {
        const bool more = (kt + 32) < K;
        if (more) {
            stage_B_async(wt + kt + 32, K, (unsigned)(size_t)&Bs[buf ^ 1][0], tid);
            wait_async2();
        } else {
            wait_async0();
        }
        __syncthreads();
        const v16bf a = load_a(ap + kt);
        const unsigned short* lb = &Bs[buf][0];
#pragma unroll
        for (int f = 0; f < 4; ++f)
            acc[f] = wmma_bf16(a, load_b(lb + (f * 16 + nl) * 32), acc[f]);
        __syncthreads();
    }
    const int rbase = (lane < 16) ? 0 : 8;
#pragma unroll
    for (int f = 0; f < 4; ++f) {
        const int col = n0 + f * 16 + nl;
        const float bv = bias[col];
#pragma unroll
        for (int r = 0; r < 8; ++r)
            C[(size_t)(m0 + rbase + r) * N + col] = f2bf(acc[f][r] + bv);
    }
}

// C[M,N](f32) = A @ W^T + bias + X   (attention out-proj + residual)
__global__ void __launch_bounds__(128) gemm_res_f32_kernel(const unsigned short* __restrict__ A,
                                                           const unsigned short* __restrict__ W,
                                                           const float* __restrict__ bias,
                                                           const float* __restrict__ X,
                                                           float* __restrict__ C,
                                                           int N, int K) {
    __shared__ __align__(16) unsigned short Bs[2][64 * 32];
    const int tid = threadIdx.x;
    const int wave = tid >> 5, lane = tid & 31;
    const int nl = lane & 15;
    const int m0 = blockIdx.y * 64 + wave * 16;
    const int n0 = blockIdx.x * 64;
    const unsigned short* ap = A + (size_t)(m0 + nl) * K;
    const unsigned short* wt = W + (size_t)n0 * K;
    v8f acc[4] = {};
    stage_B_async(wt, K, (unsigned)(size_t)&Bs[0][0], tid);
    int buf = 0;
    for (int kt = 0; kt < K; kt += 32, buf ^= 1) {
        const bool more = (kt + 32) < K;
        if (more) {
            stage_B_async(wt + kt + 32, K, (unsigned)(size_t)&Bs[buf ^ 1][0], tid);
            wait_async2();
        } else {
            wait_async0();
        }
        __syncthreads();
        const v16bf a = load_a(ap + kt);
        const unsigned short* lb = &Bs[buf][0];
#pragma unroll
        for (int f = 0; f < 4; ++f)
            acc[f] = wmma_bf16(a, load_b(lb + (f * 16 + nl) * 32), acc[f]);
        __syncthreads();
    }
    const int rbase = (lane < 16) ? 0 : 8;
#pragma unroll
    for (int f = 0; f < 4; ++f) {
        const int col = n0 + f * 16 + nl;
        const float bv = bias[col];
#pragma unroll
        for (int r = 0; r < 8; ++r) {
            size_t idx = (size_t)(m0 + rbase + r) * N + col;
            C[idx] = acc[f][r] + bv + X[idx];
        }
    }
}

// ---------------- flash attention (one wave per (head, 16-row tile)) ----------------
__global__ void __launch_bounds__(32) attn_kernel(const unsigned short* __restrict__ qb,
                                                  const unsigned short* __restrict__ kb,
                                                  const unsigned short* __restrict__ vt,
                                                  unsigned short* __restrict__ ob) {
    const int i0 = blockIdx.x * 16;
    const int h = blockIdx.y;
    const int lane = threadIdx.x & 31;
    const int nl = lane & 15;
    const int half = (lane < 16) ? 0 : 1;
    __shared__ __align__(16) unsigned short P[16 * 32];

    const unsigned short* qp = qb + (size_t)(i0 + nl) * kD + h * kDH;
    const v16bf qf0 = load_a(qp);
    const v16bf qf1 = load_a(qp + 32);

    v8f o0 = {}, o1 = {}, o2 = {}, o3 = {};
    float mrow[8], lrow[8];
#pragma unroll
    for (int r = 0; r < 8; ++r) { mrow[r] = -1e30f; lrow[r] = 0.f; }

    for (int j0 = 0; j0 <= i0 + 15; j0 += 32) {
        v8f s0 = {}, s1 = {};
        const unsigned short* kp0 = kb + (size_t)(j0 + nl) * kD + h * kDH;
        const unsigned short* kp1 = kb + (size_t)(j0 + 16 + nl) * kD + h * kDH;
        s0 = wmma_bf16(qf0, load_b(kp0), s0);
        s0 = wmma_bf16(qf1, load_b(kp0 + 32), s0);
        s1 = wmma_bf16(qf0, load_b(kp1), s1);
        s1 = wmma_bf16(qf1, load_b(kp1 + 32), s1);

#pragma unroll
        for (int r = 0; r < 8; ++r) {
            const int row = i0 + r + half * 8;
            const int t0 = j0 + nl, t1 = j0 + 16 + nl;
            float a0 = s0[r] * 0.125f + ((t0 <= row) ? 0.f : -1e9f);
            float a1 = s1[r] * 0.125f + ((t1 <= row) ? 0.f : -1e9f);
            float mx = fmaxf(a0, a1);
#pragma unroll
            for (int off = 1; off < 16; off <<= 1) mx = fmaxf(mx, __shfl_xor(mx, off, 16));
            const float nm = fmaxf(mrow[r], mx);
            const float sc = __expf(mrow[r] - nm);
            const float p0 = __expf(a0 - nm);
            const float p1 = __expf(a1 - nm);
            float rs = p0 + p1;
#pragma unroll
            for (int off = 1; off < 16; off <<= 1) rs += __shfl_xor(rs, off, 16);
            lrow[r] = lrow[r] * sc + rs;
            mrow[r] = nm;
            o0[r] *= sc; o1[r] *= sc; o2[r] *= sc; o3[r] *= sc;
            P[(r + half * 8) * 32 + nl] = f2bf(p0);
            P[(r + half * 8) * 32 + 16 + nl] = f2bf(p1);
        }
        __syncthreads();
        const v16bf pf = load_a(&P[nl * 32]);
        const unsigned short* vbase = vt + (size_t)(h * kDH + nl) * kS + j0;
        o0 = wmma_bf16(pf, load_b(vbase + (size_t)0 * 16 * kS), o0);
        o1 = wmma_bf16(pf, load_b(vbase + (size_t)1 * 16 * kS), o1);
        o2 = wmma_bf16(pf, load_b(vbase + (size_t)2 * 16 * kS), o2);
        o3 = wmma_bf16(pf, load_b(vbase + (size_t)3 * 16 * kS), o3);
        __syncthreads();
    }

#pragma unroll
    for (int r = 0; r < 8; ++r) {
        const int row = i0 + r + half * 8;
        const float inv = 1.f / lrow[r];
        unsigned short* op = ob + (size_t)row * kD + h * kDH + nl;
        op[0]  = f2bf(o0[r] * inv);
        op[16] = f2bf(o1[r] * inv);
        op[32] = f2bf(o2[r] * inv);
        op[48] = f2bf(o3[r] * inv);
    }
}

// ---------------- router / top-1 routing ----------------
__global__ void __launch_bounds__(32) router_kernel(const unsigned short* __restrict__ h2,
                                                    const float* __restrict__ rw,
                                                    const float* __restrict__ rb,
                                                    float* __restrict__ gate,
                                                    int* __restrict__ route) {
    const int t = blockIdx.x;
    const int lane = threadIdx.x & 31;
    float acc[kE];
#pragma unroll
    for (int e = 0; e < kE; ++e) acc[e] = 0.f;
    for (int d = lane; d < kD; d += 32) {
        const float hv = bf2f(h2[(size_t)t * kD + d]);
#pragma unroll
        for (int e = 0; e < kE; ++e) acc[e] += hv * rw[e * kD + d];
    }
#pragma unroll
    for (int e = 0; e < kE; ++e)
#pragma unroll
        for (int off = 16; off > 0; off >>= 1) acc[e] += __shfl_xor(acc[e], off, 32);
    if (lane == 0) {
        float mx = -1e30f; int am = 0;
        float lg[kE];
#pragma unroll
        for (int e = 0; e < kE; ++e) {
            lg[e] = acc[e] + rb[e];
            if (lg[e] > mx) { mx = lg[e]; am = e; }
        }
        float ssum = 0.f;
#pragma unroll
        for (int e = 0; e < kE; ++e) ssum += __expf(lg[e] - mx);
        gate[t] = 1.f / ssum;
        route[t] = am;
    }
}

__global__ void zero_cnt_kernel(int* cnt) {
    if (threadIdx.x < kE) cnt[threadIdx.x] = 0;
}

__global__ void build_lists_kernel(const int* __restrict__ route,
                                   int* __restrict__ cnt,
                                   int* __restrict__ perm) {
    int t = blockIdx.x * blockDim.x + threadIdx.x;
    if (t >= kS) return;
    int e = route[t];
    int pos = atomicAdd(&cnt[e], 1);
    perm[e * kS + pos] = t;
}

// ---------------- grouped MoE GEMMs (gathered rows, tiled + async staged) ----------------
// hid[tok, f] = relu( h2[tok,:] @ W1[e]^T + b1[e] )
__global__ void __launch_bounds__(128) moe_ffn1_kernel(const unsigned short* __restrict__ h2,
                                                       const unsigned short* __restrict__ W1b,
                                                       const float* __restrict__ b1,
                                                       const int* __restrict__ cnt,
                                                       const int* __restrict__ perm,
                                                       unsigned short* __restrict__ hid) {
    const int e = blockIdx.z;
    const int c = cnt[e];
    if ((int)blockIdx.y * 64 >= c) return;          // whole block exits: barrier-safe
    __shared__ __align__(16) unsigned short Bs[2][64 * 32];
    const int tid = threadIdx.x;
    const int wave = tid >> 5, lane = tid & 31;
    const int nl = lane & 15;
    const int m0 = blockIdx.y * 64 + wave * 16;
    const int n0 = blockIdx.x * 64;
    const int pos = m0 + nl;
    const int posc = (pos < c) ? pos : (c - 1);
    const int tok = perm[e * kS + posc];
    const unsigned short* ap = h2 + (size_t)tok * kD;
    const unsigned short* wt = W1b + (size_t)e * kDFF * kD + (size_t)n0 * kD;
    v8f acc[4] = {};
    stage_B_async(wt, kD, (unsigned)(size_t)&Bs[0][0], tid);
    int buf = 0;
    for (int kt = 0; kt < kD; kt += 32, buf ^= 1) {
        const bool more = (kt + 32) < kD;
        if (more) {
            stage_B_async(wt + kt + 32, kD, (unsigned)(size_t)&Bs[buf ^ 1][0], tid);
            wait_async2();
        } else {
            wait_async0();
        }
        __syncthreads();
        const v16bf a = load_a(ap + kt);
        const unsigned short* lb = &Bs[buf][0];
#pragma unroll
        for (int f = 0; f < 4; ++f)
            acc[f] = wmma_bf16(a, load_b(lb + (f * 16 + nl) * 32), acc[f]);
        __syncthreads();
    }
    const int rbase = (lane < 16) ? 0 : 8;
#pragma unroll
    for (int f = 0; f < 4; ++f) {
        const int col = n0 + f * 16 + nl;
        const float bv = b1[e * kDFF + col];
#pragma unroll
        for (int r = 0; r < 8; ++r) {
            const int p = m0 + rbase + r;
            if (p < c) {
                const int tk = perm[e * kS + p];
                float v = acc[f][r] + bv;
                hid[(size_t)tk * kDFF + col] = f2bf(v > 0.f ? v : 0.f);
            }
        }
    }
}

// out[tok, d] += gate[tok] * ( hid[tok,:] @ W2[e]^T + b2[e] )
__global__ void __launch_bounds__(128) moe_ffn2_kernel(const unsigned short* __restrict__ hid,
                                                       const unsigned short* __restrict__ W2b,
                                                       const float* __restrict__ b2,
                                                       const float* __restrict__ gate,
                                                       const int* __restrict__ cnt,
                                                       const int* __restrict__ perm,
                                                       float* __restrict__ out) {
    const int e = blockIdx.z;
    const int c = cnt[e];
    if ((int)blockIdx.y * 64 >= c) return;
    __shared__ __align__(16) unsigned short Bs[2][64 * 32];
    const int tid = threadIdx.x;
    const int wave = tid >> 5, lane = tid & 31;
    const int nl = lane & 15;
    const int m0 = blockIdx.y * 64 + wave * 16;
    const int n0 = blockIdx.x * 64;
    const int pos = m0 + nl;
    const int posc = (pos < c) ? pos : (c - 1);
    const int tok = perm[e * kS + posc];
    const unsigned short* ap = hid + (size_t)tok * kDFF;
    const unsigned short* wt = W2b + (size_t)e * kD * kDFF + (size_t)n0 * kDFF;
    v8f acc[4] = {};
    stage_B_async(wt, kDFF, (unsigned)(size_t)&Bs[0][0], tid);
    int buf = 0;
    for (int kt = 0; kt < kDFF; kt += 32, buf ^= 1) {
        const bool more = (kt + 32) < kDFF;
        if (more) {
            stage_B_async(wt + kt + 32, kDFF, (unsigned)(size_t)&Bs[buf ^ 1][0], tid);
            wait_async2();
        } else {
            wait_async0();
        }
        __syncthreads();
        const v16bf a = load_a(ap + kt);
        const unsigned short* lb = &Bs[buf][0];
#pragma unroll
        for (int f = 0; f < 4; ++f)
            acc[f] = wmma_bf16(a, load_b(lb + (f * 16 + nl) * 32), acc[f]);
        __syncthreads();
    }
    const int rbase = (lane < 16) ? 0 : 8;
#pragma unroll
    for (int f = 0; f < 4; ++f) {
        const int col = n0 + f * 16 + nl;
        const float bv = b2[e * kD + col];
#pragma unroll
        for (int r = 0; r < 8; ++r) {
            const int p = m0 + rbase + r;
            if (p < c) {
                const int tk = perm[e * kS + p];
                out[(size_t)tk * kD + col] += gate[tk] * (acc[f][r] + bv);
            }
        }
    }
}

// ---------------- host side ----------------
static inline size_t align256(size_t x) { return (x + 255) & ~(size_t)255; }

extern "C" void kernel_launch(void* const* d_in, const int* in_sizes, int n_in,
                              void* d_out, int out_size, void* d_ws, size_t ws_size,
                              hipStream_t stream) {
    const float* x    = (const float*)d_in[0];
    // d_in[1] = causal_mask (unused, computed analytically)
    const float* Wk   = (const float*)d_in[2];
    const float* bk   = (const float*)d_in[3];
    const float* Wq   = (const float*)d_in[4];
    const float* bq   = (const float*)d_in[5];
    const float* Wv   = (const float*)d_in[6];
    const float* bv   = (const float*)d_in[7];
    const float* in_w = (const float*)d_in[8];
    const float* in_b = (const float*)d_in[9];
    const float* Wo   = (const float*)d_in[10];
    const float* bo   = (const float*)d_in[11];
    const float* ln1g = (const float*)d_in[12];
    const float* ln1b = (const float*)d_in[13];
    const float* ln2g = (const float*)d_in[14];
    const float* ln2b = (const float*)d_in[15];
    const float* rw   = (const float*)d_in[16];
    const float* rb   = (const float*)d_in[17];
    const float* W1   = (const float*)d_in[18];
    const float* b1   = (const float*)d_in[19];
    const float* W2   = (const float*)d_in[20];
    const float* b2   = (const float*)d_in[21];
    float* out = (float*)d_out;

    // ---- carve workspace ----
    char* p = (char*)d_ws;
    auto alloc = [&](size_t bytes) { void* r = (void*)p; p += align256(bytes); return r; };
    unsigned short* WkB = (unsigned short*)alloc((size_t)kD * kD * 2);
    unsigned short* WqB = (unsigned short*)alloc((size_t)kD * kD * 2);
    unsigned short* WvB = (unsigned short*)alloc((size_t)kD * kD * 2);
    unsigned short* WoB = (unsigned short*)alloc((size_t)kD * kD * 2);
    unsigned short* WiB = (unsigned short*)alloc((size_t)3 * kD * kD * 2);
    unsigned short* W1B = (unsigned short*)alloc((size_t)kE * kDFF * kD * 2);
    unsigned short* W2B = (unsigned short*)alloc((size_t)kE * kD * kDFF * 2);
    unsigned short* h1  = (unsigned short*)alloc((size_t)kS * kD * 2);
    unsigned short* Kb  = (unsigned short*)alloc((size_t)kS * kD * 2);
    unsigned short* Qb  = (unsigned short*)alloc((size_t)kS * kD * 2);
    unsigned short* Vb  = (unsigned short*)alloc((size_t)kS * kD * 2);
    unsigned short* qb  = (unsigned short*)alloc((size_t)kS * kD * 2);
    unsigned short* kb  = (unsigned short*)alloc((size_t)kS * kD * 2);
    unsigned short* vb  = (unsigned short*)alloc((size_t)kS * kD * 2);
    unsigned short* vt  = (unsigned short*)alloc((size_t)kS * kD * 2);
    unsigned short* ob  = (unsigned short*)alloc((size_t)kS * kD * 2);
    unsigned short* h2  = (unsigned short*)alloc((size_t)kS * kD * 2);
    unsigned short* hid = (unsigned short*)alloc((size_t)kS * kDFF * 2);
    float* x1   = (float*)alloc((size_t)kS * kD * 4);
    float* gate = (float*)alloc((size_t)kS * 4);
    int* route  = (int*)alloc((size_t)kS * 4);
    int* cnt    = (int*)alloc((size_t)kE * 4);
    int* perm   = (int*)alloc((size_t)kE * kS * 4);
    (void)ws_size; (void)n_in; (void)in_sizes; (void)out_size;

    // ---- weight conversions to bf16 ----
    auto cvt = [&](const float* src, unsigned short* dst, int n) {
        int grid = (n + 255) / 256; if (grid > 4096) grid = 4096;
        cvt_bf16_kernel<<<grid, 256, 0, stream>>>(src, dst, n);
    };
    cvt(Wk, WkB, kD * kD);
    cvt(Wq, WqB, kD * kD);
    cvt(Wv, WvB, kD * kD);
    cvt(Wo, WoB, kD * kD);
    cvt(in_w, WiB, 3 * kD * kD);
    cvt(W1, W1B, kE * kDFF * kD);
    cvt(W2, W2B, kE * kD * kDFF);

    const dim3 gemmGrid(kD / 64, kS / 64);     // (12, 32), 128 threads

    // ---- MHA block ----
    ln_bf16_kernel<<<kS, 256, 0, stream>>>(x, ln1g, ln1b, h1);
    gemm_bf16_kernel<<<gemmGrid, 128, 0, stream>>>(h1, WkB, bk, Kb, kD, kD);
    gemm_bf16_kernel<<<gemmGrid, 128, 0, stream>>>(h1, WqB, bq, Qb, kD, kD);
    gemm_bf16_kernel<<<gemmGrid, 128, 0, stream>>>(h1, WvB, bv, Vb, kD, kD);
    // torch quirk: multihead_attn(K, Q, V) -> query = in_proj_q(K), key = in_proj_k(Q)
    gemm_bf16_kernel<<<gemmGrid, 128, 0, stream>>>(Kb, WiB,                      in_b,          qb, kD, kD);
    gemm_bf16_kernel<<<gemmGrid, 128, 0, stream>>>(Qb, WiB + (size_t)kD * kD,     in_b + kD,     kb, kD, kD);
    gemm_bf16_kernel<<<gemmGrid, 128, 0, stream>>>(Vb, WiB + (size_t)2 * kD * kD, in_b + 2 * kD, vb, kD, kD);
    transpose_v_kernel<<<(kS * kD + 255) / 256, 256, 0, stream>>>(vb, vt);
    attn_kernel<<<dim3(kS / 16, kH), 32, 0, stream>>>(qb, kb, vt, ob);
    gemm_res_f32_kernel<<<gemmGrid, 128, 0, stream>>>(ob, WoB, bo, x, x1, kD, kD);

    // ---- MoE block ----
    ln_bf16_kernel<<<kS, 256, 0, stream>>>(x1, ln2g, ln2b, h2);
    router_kernel<<<kS, 32, 0, stream>>>(h2, rw, rb, gate, route);
    zero_cnt_kernel<<<1, 32, 0, stream>>>(cnt);
    build_lists_kernel<<<kS / 256, 256, 0, stream>>>(route, cnt, perm);
    copy_f32_kernel<<<4096, 256, 0, stream>>>(x1, out, kS * kD);
    moe_ffn1_kernel<<<dim3(kDFF / 64, kS / 64, kE), 128, 0, stream>>>(h2, W1B, b1, cnt, perm, hid);
    moe_ffn2_kernel<<<dim3(kD / 64, kS / 64, kE), 128, 0, stream>>>(hid, W2B, b2, gate, cnt, perm, out);
}
